// LayerNormDense_4870492913836
// MI455X (gfx1250) — 8-bit matrix operand order repaired and checked by reading, compiled, not run on hardware
//
#include <hip/hip_runtime.h>


typedef __attribute__((ext_vector_type(16))) int   v16i;
typedef __attribute__((ext_vector_type(8)))  float v8f;
typedef __attribute__((ext_vector_type(4)))  float f4v;

typedef int v4i_t __attribute__((vector_size(4 * sizeof(int))));
typedef __attribute__((address_space(1))) v4i_t* g_v4i_p;
typedef __attribute__((address_space(3))) v4i_t* l_v4i_p;

#define TOKENS 65536
#define HIDDEN 768
#define UNITS  768
#define KTILES 6
#define NT_TOT 48
#define LDS_PITCH 136

#if __has_builtin(__builtin_amdgcn_global_load_async_to_lds_b128)
#define HAVE_ASYNC_LDS 1
#else
#define HAVE_ASYNC_LDS 0
#endif

__device__ __forceinline__ void wait_async0() {
#if __has_builtin(__builtin_amdgcn_s_wait_asynccnt)
  __builtin_amdgcn_s_wait_asynccnt(0);
#else
  asm volatile("s_wait_asynccnt 0x0" ::: "memory");
#endif
}

__device__ __forceinline__ unsigned f32_to_e4m3_byte(float f) {
  unsigned u = __float_as_uint(f);
  unsigned s = (u >> 24) & 0x80u;
  unsigned a = u & 0x7FFFFFFFu;
  if (a >= 0x7F800000u) return s | 0x7Fu;
  if (a >= 0x43E80000u) return s | 0x7Eu;
  int e = (int)(a >> 23) - 127;
  unsigned mant = (a & 0x7FFFFFu) | 0x800000u;
  int shift = 20;
  if (e < -6) { shift += (-6 - e); e = -6; }
  if (shift > 24) return s;
  unsigned keep = mant >> shift;
  unsigned rem  = mant & ((1u << shift) - 1u);
  unsigned half = 1u << (shift - 1);
  keep += (unsigned)((rem > half) || ((rem == half) && (keep & 1u)));
  if (keep >= 16u) { keep >>= 1; e += 1; }
  if (e > 8) return s | 0x7Eu;
  if (keep < 8u) return s | keep;
  return s | ((unsigned)(e + 7) << 3) | (keep & 7u);
}

__device__ __forceinline__ unsigned pack4_e4m3(float f0, float f1, float f2, float f3) {
#if __has_builtin(__builtin_amdgcn_cvt_pk_fp8_f32)
  int r = 0;
  r = __builtin_amdgcn_cvt_pk_fp8_f32(f0, f1, r, false);
  r = __builtin_amdgcn_cvt_pk_fp8_f32(f2, f3, r, true);
  return (unsigned)r;
#else
  return f32_to_e4m3_byte(f0)        | (f32_to_e4m3_byte(f1) << 8) |
         (f32_to_e4m3_byte(f2) << 16)| (f32_to_e4m3_byte(f3) << 24);
#endif
}

__global__ __launch_bounds__(256) void ln_quant_kernel(
    const float* __restrict__ x, const float* __restrict__ gamma,
    const float* __restrict__ beta, unsigned* __restrict__ aq) {
  const int wave = threadIdx.x >> 5;
  const int lane = threadIdx.x & 31;
  const int row  = blockIdx.x * 8 + wave;

  const f4v* xr = (const f4v*)(x + (size_t)row * HIDDEN);
  const f4v* g4 = (const f4v*)gamma;
  const f4v* b4 = (const f4v*)beta;

  f4v v[6];
  float sum = 0.f, ssq = 0.f;
#pragma unroll
  for (int j = 0; j < 6; ++j) {
    f4v t = __builtin_nontemporal_load(&xr[lane + 32 * j]);
    v[j] = t;
    sum += t.x + t.y + t.z + t.w;
    ssq += t.x * t.x + t.y * t.y + t.z * t.z + t.w * t.w;
  }
#pragma unroll
  for (int off = 16; off >= 1; off >>= 1) {
    sum += __shfl_xor(sum, off, 32);
    ssq += __shfl_xor(ssq, off, 32);
  }
  const float mu  = sum * (1.0f / HIDDEN);
  const float var = ssq * (1.0f / HIDDEN) - mu * mu;
  const float rs  = rsqrtf(var + 1e-5f);

  unsigned* ar = aq + (size_t)row * (HIDDEN / 4);
#pragma unroll
  for (int j = 0; j < 6; ++j) {
    const int c4 = lane + 32 * j;
    const f4v gg = g4[c4];
    const f4v bb = b4[c4];
    const f4v t  = v[j];
    ar[c4] = pack4_e4m3((t.x - mu) * rs * gg.x + bb.x,
                        (t.y - mu) * rs * gg.y + bb.y,
                        (t.z - mu) * rs * gg.z + bb.z,
                        (t.w - mu) * rs * gg.w + bb.w);
  }
}

__global__ __launch_bounds__(256) void wq_swizzle_kernel(
    const float* __restrict__ w, unsigned* __restrict__ bsw) {
  const int d    = blockIdx.x * 256 + threadIdx.x;
  const int v    = d & 15;
  const int lane = (d >> 4) & 31;
  const int kt   = (d >> 9) % KTILES;
  const int nt   = d / (16 * 32 * KTILES);

  const int n    = nt * 16 + (lane & 15);
  const int half = lane >> 4;
  const int kb   = kt * 128 + (v >> 1) * 16 + half * 8 + (v & 1) * 4;

  const float* col = w + n;
  bsw[d] = pack4_e4m3(col[(size_t)(kb + 0) * UNITS],
                      col[(size_t)(kb + 1) * UNITS],
                      col[(size_t)(kb + 2) * UNITS],
                      col[(size_t)(kb + 3) * UNITS]);
}

__global__ __launch_bounds__(256) void fp8_gemm_kernel(
    const unsigned char* __restrict__ aq, const unsigned* __restrict__ bsw,
    const float* __restrict__ bias, float* __restrict__ out) {
  __shared__ unsigned char smA[2][128 * LDS_PITCH];

  const int tid  = threadIdx.x;
  const int wave = tid >> 5;
  const int lane = tid & 31;
  const int m0   = blockIdx.x * 128;
  const int n0   = blockIdx.y * 128;

  int sr[4], sc[4];
#pragma unroll
  for (int i = 0; i < 4; ++i) {
    const int c = tid + 256 * i;
    sr[i] = c >> 3;
    sc[i] = (c & 7) * 16;
  }

  v8f acc[8];
#pragma unroll
  for (int t = 0; t < 8; ++t) {
#pragma unroll
    for (int i = 0; i < 8; ++i) acc[t][i] = 0.0f;
  }

  const int rowInTile = wave * 16 + (lane & 15);
  const int chunkoff  = (lane >> 4) * 8;

#if HAVE_ASYNC_LDS
#pragma unroll
  for (int i = 0; i < 4; ++i)
    __builtin_amdgcn_global_load_async_to_lds_b128(
        (g_v4i_p)(aq + (size_t)(m0 + sr[i]) * HIDDEN + sc[i]),
        (l_v4i_p)&smA[0][sr[i] * LDS_PITCH + sc[i]], 0, 0);
  wait_async0();
  __syncthreads();
#else
  {
    uint4 st[4];
#pragma unroll
    for (int i = 0; i < 4; ++i)
      st[i] = *(const uint4*)(aq + (size_t)(m0 + sr[i]) * HIDDEN + sc[i]);
#pragma unroll
    for (int i = 0; i < 4; ++i) {
      unsigned long long* dst = (unsigned long long*)&smA[0][sr[i] * LDS_PITCH + sc[i]];
      dst[0] = ((const unsigned long long*)&st[i])[0];
      dst[1] = ((const unsigned long long*)&st[i])[1];
    }
    __syncthreads();
  }
#endif

  for (int kt = 0; kt < KTILES; ++kt) {
    const int cur = kt & 1;
    const int nxt = cur ^ 1;

#if HAVE_ASYNC_LDS
    if (kt + 1 < KTILES) {
#pragma unroll
      for (int i = 0; i < 4; ++i)
        __builtin_amdgcn_global_load_async_to_lds_b128(
            (g_v4i_p)(aq + (size_t)(m0 + sr[i]) * HIDDEN + (kt + 1) * 128 + sc[i]),
            (l_v4i_p)&smA[nxt][sr[i] * LDS_PITCH + sc[i]], 0, 0);
    }
#else
    uint4 st[4];
    if (kt + 1 < KTILES) {
#pragma unroll
      for (int i = 0; i < 4; ++i)
        st[i] = *(const uint4*)(aq + (size_t)(m0 + sr[i]) * HIDDEN + (kt + 1) * 128 + sc[i]);
    }
#endif

    union { v16i v; unsigned long long d[8]; } af;
#pragma unroll
    for (int i = 0; i < 8; ++i)
      af.d[i] = *(const unsigned long long*)&smA[cur][rowInTile * LDS_PITCH + chunkoff + i * 16];

    const uint4* bbase = (const uint4*)(bsw +
        ((((size_t)(blockIdx.y * 8)) * KTILES + kt) * 32 + lane) * 16);
    union BF { v16i v; uint4 q[4]; } bf0, bf1;
#pragma unroll
    for (int i = 0; i < 4; ++i) bf0.q[i] = bbase[i];

#pragma unroll
    for (int nt = 0; nt < 8; ++nt) {
      if ((nt & 1) == 0) {
        if (nt + 1 < 8) {
          const uint4* bp = bbase + (size_t)(nt + 1) * (KTILES * 32 * 4);
#pragma unroll
          for (int i = 0; i < 4; ++i) bf1.q[i] = bp[i];
        }
        acc[nt] = __builtin_amdgcn_wmma_f32_16x16x128_fp8_fp8(
            af.v, bf0.v, (short)0, acc[nt], false, false);
      } else {
        if (nt + 1 < 8) {
          const uint4* bp = bbase + (size_t)(nt + 1) * (KTILES * 32 * 4);
#pragma unroll
          for (int i = 0; i < 4; ++i) bf0.q[i] = bp[i];
        }
        acc[nt] = __builtin_amdgcn_wmma_f32_16x16x128_fp8_fp8(
            af.v, bf1.v, (short)0, acc[nt], false, false);
      }
    }

    if (kt + 1 < KTILES) {
#if HAVE_ASYNC_LDS
      wait_async0();
#else
#pragma unroll
      for (int i = 0; i < 4; ++i) {
        unsigned long long* dst = (unsigned long long*)&smA[nxt][sr[i] * LDS_PITCH + sc[i]];
        dst[0] = ((const unsigned long long*)&st[i])[0];
        dst[1] = ((const unsigned long long*)&st[i])[1];
      }
#endif
      __syncthreads();
    }
  }

  const int rbase = m0 + wave * 16 + (lane >> 4) * 8;
#pragma unroll
  for (int nt = 0; nt < 8; ++nt) {
    const int col = n0 + nt * 16 + (lane & 15);
    const float bv = bias[col];
#pragma unroll
    for (int g = 0; g < 8; ++g) {
      __builtin_nontemporal_store(acc[nt][g] + bv,
                                  &out[(size_t)(rbase + g) * UNITS + col]);
    }
  }
}

extern "C" void kernel_launch(void* const* d_in, const int* in_sizes, int n_in,
                              void* d_out, int out_size, void* d_ws, size_t ws_size,
                              hipStream_t stream) {
  (void)in_sizes; (void)n_in; (void)out_size; (void)ws_size;
  const float* x      = (const float*)d_in[0];
  const float* gamma  = (const float*)d_in[1];
  const float* beta   = (const float*)d_in[2];
  const float* kernel = (const float*)d_in[3];
  const float* bias   = (const float*)d_in[4];
  float* out = (float*)d_out;

  unsigned char* aq  = (unsigned char*)d_ws;
  unsigned*      bsw = (unsigned*)(aq + (size_t)TOKENS * HIDDEN);

  ln_quant_kernel<<<TOKENS / 8, 256, 0, stream>>>(x, gamma, beta, (unsigned*)aq);
  wq_swizzle_kernel<<<(NT_TOT * KTILES * 32 * 16) / 256, 256, 0, stream>>>(kernel, bsw);
  fp8_gemm_kernel<<<dim3(TOKENS / 128, UNITS / 128), 256, 0, stream>>>(aq, bsw, bias, out);
}
